// Conv4dNet_44796508897564
// MI455X (gfx1250) — matrix operand order repaired and checked by reading, not compiled, not run on hardware
//
#include <hip/hip_runtime.h>

// ---------------- CDNA5 WMMA types ----------------
typedef __attribute__((ext_vector_type(16))) __bf16        v16bf;
typedef __attribute__((ext_vector_type(8)))  float         v8f;
typedef __attribute__((ext_vector_type(8)))  unsigned int  v8u;

// ---------------- problem geometry ----------------
constexpr int DD   = 16;                 // spatial extent per dim
constexpr int PD   = 18;                 // padded extent (pad=1 both sides)
constexpr int ST3  = PD;
constexpr int ST2  = PD * PD;
constexpr int ST1  = PD * PD * PD;
constexpr int SC   = PD * PD * PD * PD;  // channel stride = 104976
constexpr int VOL  = DD * DD * DD * DD;  // 65536
constexpr int BATCH = 2;

// K padding (multiple of 32)
constexpr int KP1 = 96;     // CIN=1  -> K=81
constexpr int KP2 = 2592;   // CIN=32
constexpr int KP3 = 5184;   // CIN=64
constexpr int KP4 = 2592;   // CIN=32

// ---------------- workspace layout ----------------
constexpr size_t alignup(size_t x) { return (x + 4095) & ~(size_t)4095; }
constexpr size_t BYTES_A0 = (size_t)BATCH * 1  * SC * 2;
constexpr size_t BYTES_A1 = (size_t)BATCH * 32 * SC * 2;
constexpr size_t BYTES_A2 = (size_t)BATCH * 64 * SC * 2;
constexpr size_t BYTES_A3 = (size_t)BATCH * 32 * SC * 2;
constexpr size_t BYTES_W1 = (size_t)32 * KP1 * 2;
constexpr size_t BYTES_W2 = (size_t)64 * KP2 * 2;
constexpr size_t BYTES_W3 = (size_t)32 * KP3 * 2;
constexpr size_t BYTES_W4 = (size_t)16 * KP4 * 2;   // COUT=1 padded to 16 rows (zeros)
constexpr size_t OFF_A0 = 0;
constexpr size_t OFF_A1 = alignup(OFF_A0 + BYTES_A0);
constexpr size_t OFF_A2 = alignup(OFF_A1 + BYTES_A1);
constexpr size_t OFF_A3 = alignup(OFF_A2 + BYTES_A2);
constexpr size_t OFF_W1 = alignup(OFF_A3 + BYTES_A3);
constexpr size_t OFF_W2 = alignup(OFF_W1 + BYTES_W1);
constexpr size_t OFF_W3 = alignup(OFF_W2 + BYTES_W2);
constexpr size_t OFF_W4 = alignup(OFF_W3 + BYTES_W3);
constexpr size_t WS_END = alignup(OFF_W4 + BYTES_W4);   // ~55 MB

// ---------------- utility kernels ----------------
__global__ void zero_ws_kernel(uint4* __restrict__ p, size_t n16) {
    size_t i = (size_t)blockIdx.x * blockDim.x + threadIdx.x;
    size_t stride = (size_t)gridDim.x * blockDim.x;
    uint4 z; z.x = 0u; z.y = 0u; z.z = 0u; z.w = 0u;
    for (; i < n16; i += stride) p[i] = z;
}

// x f32 [2,1,16^4] -> padded bf16 [2,1,18^4] interior
__global__ void pad_input_kernel(const float* __restrict__ x, __bf16* __restrict__ a0) {
    int i = blockIdx.x * blockDim.x + threadIdx.x;
    if (i >= BATCH * VOL) return;
    int b  = i >> 16;
    int r  = i & (VOL - 1);
    int d1 = r >> 12, d2 = (r >> 8) & 15, d3 = (r >> 4) & 15, d4 = r & 15;
    a0[(size_t)b * SC + (size_t)(d1 + 1) * ST1 + (d2 + 1) * ST2 + (d3 + 1) * ST3 + (d4 + 1)]
        = (__bf16)x[i];
}

// Swizzle f32 weights [COUT, K] into per-lane A-fragment order (bf16):
// dst[(((co*NCH + kc)*2 + g)*16 + e], where k = kc*32 + 8g + e + (e>=8 ? 8 : 0).
__global__ void wprep_kernel(const float* __restrict__ w, __bf16* __restrict__ dst,
                             int cout, int K, int nch) {
    int i = blockIdx.x * blockDim.x + threadIdx.x;
    if (i >= cout * K) return;
    int co = i / K;
    int k  = i % K;
    int kc = k >> 5, kl = k & 31;
    int g, e;
    if (kl < 8)       { g = 0; e = kl; }
    else if (kl < 16) { g = 1; e = kl - 8; }
    else if (kl < 24) { g = 0; e = kl - 8; }
    else              { g = 1; e = kl - 16; }
    dst[(((size_t)co * nch + kc) * 2 + g) * 16 + e] = (__bf16)w[i];
}

// ---------------- implicit-GEMM conv4d layer ----------------
// D(16co x 16pos) += A(weights 16x32 bf16) x B(activations 32x16 bf16), f32 acc.
template <int CIN, int COUT, int KPAD, bool RELU, bool FINAL>
__global__ void __launch_bounds__(256)
conv4d_wmma_kernel(const __bf16* __restrict__ inp,
                   const __bf16* __restrict__ wsw,
                   const float*  __restrict__ bias,
                   __bf16* __restrict__ outp,
                   float*  __restrict__ outf) {
    constexpr int K   = CIN * 81;
    constexpr int NCH = KPAD / 32;
    constexpr int NT  = (COUT + 15) / 16;

    __shared__ __align__(16) int aoffB[KPAD];   // per-k byte offset into padded input
    __shared__ float sbias[NT * 16];

    const int tid = threadIdx.x;

    // Per-k gather byte-offsets, built once. k >= K -> offset 0 (valid address;
    // contribution is zeroed by the zero-padded weights).
    for (int k = tid; k < KPAD; k += 256) {
        int off = 0;
        if (k < K) {
            int ci = k / 81;
            int t  = k % 81;
            int o1 = t / 27, o2 = (t / 9) % 3, o3 = (t / 3) % 3, o4 = t % 3;
            off = 2 * (ci * SC + o1 * ST1 + o2 * ST2 + o3 * ST3 + o4);
        }
        aoffB[k] = off;
    }
    if (tid < NT * 16) sbias[tid] = (tid < COUT) ? bias[tid] : 0.0f;
    __syncthreads();   // only barrier; waves independent afterwards

    const int wave = tid >> 5;
    const int lane = tid & 31;
    const int g    = lane >> 4;   // lane half
    const int mn   = lane & 15;   // B/D column: position d4; A/D row: cout lane

    const int tileId = blockIdx.x * 8 + wave;   // (b, d1, d2, d3)
    const int b  = tileId >> 12;
    const int d1 = (tileId >> 8) & 15;
    const int d2 = (tileId >> 4) & 15;
    const int d3 = tileId & 15;

    const int tileBase = b * (CIN * SC) + d1 * ST1 + d2 * ST2 + d3 * ST3;
    const int tbn2 = 2 * (tileBase + mn);       // byte offset of this lane's column
    const char* ib = (const char*)inp;          // SGPR base + 32-bit voffset

    v8f zero = {};
    v8f acc[NT];
#pragma unroll
    for (int nt = 0; nt < NT; ++nt) acc[nt] = zero;

    for (int kc = 0; kc < NCH; ++kc) {
        // 16 gather offsets for this lane half: k = kc*32 + 8g + e + (e>=8 ? 8 : 0)
        const int4* ap = (const int4*)&aoffB[kc * 32 + (g << 3)];
        const int4* ap2 = (const int4*)&aoffB[kc * 32 + 16 + (g << 3)];
        int4 q0 = ap[0], q1 = ap[1], q2 = ap2[0], q3 = ap2[1];
        int offs[16] = { q0.x, q0.y, q0.z, q0.w, q1.x, q1.y, q1.z, q1.w,
                         q2.x, q2.y, q2.z, q2.w, q3.x, q3.y, q3.z, q3.w };

        // B fragment: activations. Build each 32-bit register pair as
        // lo | (hi<<16) so the backend uses global_load_d16(_hi)_b16 writes
        // straight into the packed VGPR halves (no v_mov_b16 repacking).
        v8u bw;
#pragma unroll
        for (int i = 0; i < 8; ++i) {
            unsigned lo = *(const unsigned short*)(ib + (unsigned)(tbn2 + offs[2 * i]));
            unsigned hi = *(const unsigned short*)(ib + (unsigned)(tbn2 + offs[2 * i + 1]));
            bw[i] = lo | (hi << 16);
        }
        v16bf bfrag = __builtin_bit_cast(v16bf, bw);

        // A fragments: pre-swizzled weights, one aligned 32B load per Cout tile;
        // back-to-back WMMAs reuse the (expensive) activation fragment.
#pragma unroll
        for (int nt = 0; nt < NT; ++nt) {
            const v16bf* af = (const v16bf*)
                (wsw + ((((size_t)(nt * 16 + mn)) * NCH + kc) * 2 + g) * 16);
            acc[nt] = __builtin_amdgcn_wmma_f32_16x16x32_bf16(
                false, *af, false, bfrag, (short)0, acc[nt], false, false);
        }
    }

    // Epilogue. D layout: lane L, vgpr r -> M(co) = r + 8g, N(pos d4) = L%16.
    if constexpr (FINAL) {
        if (g == 0) {   // only co == 0 exists
            size_t base = (size_t)b * VOL + d1 * 4096 + d2 * 256 + d3 * 16;
            outf[base + mn] = acc[0][0] + sbias[0];
        }
    } else {
        __bf16* ob = outp + (size_t)b * ((size_t)COUT * SC)
                   + (size_t)(d1 + 1) * ST1 + (d2 + 1) * ST2 + (d3 + 1) * ST3
                   + (mn + 1);
#pragma unroll
        for (int nt = 0; nt < NT; ++nt) {
#pragma unroll
            for (int r = 0; r < 8; ++r) {
                int co = nt * 16 + r + (g << 3);
                float v = acc[nt][r] + sbias[co];
                if (RELU) v = v > 0.0f ? v : 0.0f;
                ob[(size_t)co * SC] = (__bf16)v;
            }
        }
    }
}

// ---------------- host launcher ----------------
extern "C" void kernel_launch(void* const* d_in, const int* in_sizes, int n_in,
                              void* d_out, int out_size, void* d_ws, size_t ws_size,
                              hipStream_t stream) {
    (void)in_sizes; (void)n_in; (void)out_size;
    if (ws_size < WS_END) return;   // constant per run; deterministic

    const float* x  = (const float*)d_in[0];
    const float* w1 = (const float*)d_in[1];
    const float* b1 = (const float*)d_in[2];
    const float* w2 = (const float*)d_in[3];
    const float* b2 = (const float*)d_in[4];
    const float* w3 = (const float*)d_in[5];
    const float* b3 = (const float*)d_in[6];
    const float* w4 = (const float*)d_in[7];
    const float* b4 = (const float*)d_in[8];

    char* ws = (char*)d_ws;
    __bf16* A0 = (__bf16*)(ws + OFF_A0);
    __bf16* A1 = (__bf16*)(ws + OFF_A1);
    __bf16* A2 = (__bf16*)(ws + OFF_A2);
    __bf16* A3 = (__bf16*)(ws + OFF_A3);
    __bf16* W1 = (__bf16*)(ws + OFF_W1);
    __bf16* W2 = (__bf16*)(ws + OFF_W2);
    __bf16* W3 = (__bf16*)(ws + OFF_W3);
    __bf16* W4 = (__bf16*)(ws + OFF_W4);

    // 1. zero workspace (activation borders + weight K/row padding must be 0)
    zero_ws_kernel<<<4096, 256, 0, stream>>>((uint4*)ws, WS_END / 16);

    // 2. pad + convert input, swizzle weights to bf16 A-fragment order
    pad_input_kernel<<<(BATCH * VOL + 255) / 256, 256, 0, stream>>>(x, A0);
    wprep_kernel<<<(32 * 81   + 255) / 256, 256, 0, stream>>>(w1, W1, 32, 1  * 81, KP1 / 32);
    wprep_kernel<<<(64 * 2592 + 255) / 256, 256, 0, stream>>>(w2, W2, 64, 32 * 81, KP2 / 32);
    wprep_kernel<<<(32 * 5184 + 255) / 256, 256, 0, stream>>>(w3, W3, 32, 64 * 81, KP3 / 32);
    wprep_kernel<<<(1  * 2592 + 255) / 256, 256, 0, stream>>>(w4, W4, 1,  32 * 81, KP4 / 32);

    // 3. four conv layers: implicit GEMM on v_wmma_f32_16x16x32_bf16
    conv4d_wmma_kernel<1, 32, KP1, true, false>
        <<<dim3(1024), 256, 0, stream>>>(A0, W1, b1, A1, nullptr);
    conv4d_wmma_kernel<32, 64, KP2, true, false>
        <<<dim3(1024), 256, 0, stream>>>(A1, W2, b2, A2, nullptr);
    conv4d_wmma_kernel<64, 32, KP3, true, false>
        <<<dim3(1024), 256, 0, stream>>>(A2, W3, b3, A3, nullptr);
    conv4d_wmma_kernel<32, 1, KP4, false, true>
        <<<dim3(1024), 256, 0, stream>>>(A3, W4, b4, nullptr, (float*)d_out);
}
